// DGMNet_82583631168044
// MI455X (gfx1250) — compile-verified, matrix operand order checked by reading, not run on hardware
//
#include <hip/hip_runtime.h>

#define BROWS 128      // batch rows per block
#define TPB   256      // 8 waves of 32
#define HIDN  128
#define XDIM  100

typedef __attribute__((ext_vector_type(16))) _Float16 v16h;
typedef __attribute__((ext_vector_type(8)))  _Float16 h8;
typedef __attribute__((ext_vector_type(8)))  float    v8f;

__device__ __forceinline__ v8f wmma_f16(v16h a, v16h b, v8f c) {
  // D = A(16x32 f16) * B(32x16 f16) + C(16x16 f32)
  return __builtin_amdgcn_wmma_f32_16x16x32_f16(false, a, false, b, (short)0, c, false, false);
}

// tanh(x) = 1 - 2/(e^{2x}+1); exp2 overflow -> +1, underflow -> -1 (no guards needed).
// 5 VALU ops, 2 of them TRANS (co-execute with WMMA).
__device__ __forceinline__ float fast_tanh(float x) {
  float e  = __builtin_amdgcn_exp2f(x * 2.88539008177792681472f);  // e^{2x}
  float rc = __builtin_amdgcn_rcpf(e + 1.0f);
  return __builtin_fmaf(-2.0f, rc, 1.0f);
}

// Load one A-fragment (16 rows x 32 K-slice) from a row-major f16 buffer.
// srow points at this lane's row (m = lane&15), g = lane>>4 selects K-halves
// per the ISA 16-bit A-matrix layout (VGPR0..3: K=0..7 / 8..15 ; VGPR4..7: +16).
__device__ __forceinline__ v16h load_afrag(const _Float16* srow, int ks, int g) {
  const _Float16* p = srow + ks * 32 + g * 8;
  h8 lo = *(const h8*)p;          // halves 0..7  -> K = 32ks + 8g + 0..7
  h8 hi = *(const h8*)(p + 16);   // halves 8..15 -> K = 32ks + 16 + 8g + 0..7
  v16h f;
#pragma unroll
  for (int i = 0; i < 8; ++i) { f[i] = lo[i]; f[8 + i] = hi[i]; }
  return f;
}

// Stage a [K x 128] f32 row-major weight into LDS as pre-swizzled WMMA
// B-fragments: frag (ks,nt) stored as 32 lanes x 16 halves contiguous.
// Row Kvalid gets the (combined) bias vector (pairs with the constant-1
// column appended to xt); rows beyond are zero (A side zero-padded too).
__device__ __forceinline__ void stage_weight(_Float16* dst, const float* __restrict__ W,
                                             int Kvalid, const float* __restrict__ b1,
                                             const float* __restrict__ b2, int tid) {
  for (int idx = tid; idx < HIDN * HIDN; idx += TPB) {
    int k = idx >> 7, n = idx & 127;
    float v = 0.0f;
    if (k < Kvalid)            v = W[k * HIDN + n];
    else if (k == Kvalid && b1) v = b1[n] + (b2 ? b2[n] : 0.0f);
    int ks = k >> 5, kr = k & 31;
    int g  = (kr >> 3) & 1;
    int h  = (kr & 7) | ((kr >> 4) << 3);
    int ln = (n & 15) | (g << 4);
    int nt = n >> 4;
    dst[(((ks << 3) | nt) * 32 + ln) * 16 + h] = (_Float16)v;
  }
}

__global__ void __launch_bounds__(TPB)
dgm_fused_kernel(const float* __restrict__ x, const float* __restrict__ t,
                 const float* __restrict__ Sw_w, const float* __restrict__ Sw_b,
                 const float* __restrict__ Uz_w, const float* __restrict__ Uz_b,
                 const float* __restrict__ Wsz_w, const float* __restrict__ Wsz_b,
                 const float* __restrict__ Ug_w, const float* __restrict__ Ug_b,
                 const float* __restrict__ Wsg_w, const float* __restrict__ Wsg_b,
                 const float* __restrict__ Ur_w, const float* __restrict__ Ur_b,
                 const float* __restrict__ Wsr_w, const float* __restrict__ Wsr_b,
                 const float* __restrict__ Uh_w, const float* __restrict__ Uh_b,
                 const float* __restrict__ Wsh_w, const float* __restrict__ Wsh_b,
                 const float* __restrict__ Wf_w, const float* __restrict__ Wf_b,
                 const int* __restrict__ nlp, float* __restrict__ out)
{
  extern __shared__ __align__(32) char smem[];
  _Float16* wbuf  = (_Float16*)smem;                // 5 slots x 16384 halves (163840 B)
  _Float16* srbuf = (_Float16*)(smem + 131072);     // S*R buffer, overlaps slot 4 (phase 2 only)
  _Float16* sbuf  = (_Float16*)(smem + 163840);     // xt / S tile [128][128] f16 (32768 B)
  _Float16* wfh   = (_Float16*)(smem + 196608);     // Wf column as f16 (128)

  const int tid  = threadIdx.x;
  const int wave = tid >> 5, lane = tid & 31;
  const int ln   = lane & 15, g = lane >> 4;
  const int rowBase = blockIdx.x * BROWS;

  if (tid < HIDN) wfh[tid] = (_Float16)Wf_w[tid];

  // ---- stage phase-1 weights (K=101 valid, row 101 = fused biases) ----
  stage_weight(wbuf + 0 * 16384, Sw_w, 101, Sw_b, nullptr, tid);
  stage_weight(wbuf + 1 * 16384, Uz_w, 101, Uz_b, Wsz_b, tid);
  stage_weight(wbuf + 2 * 16384, Ug_w, 101, Ug_b, Wsg_b, tid);
  stage_weight(wbuf + 3 * 16384, Ur_w, 101, Ur_b, Wsr_b, tid);
  stage_weight(wbuf + 4 * 16384, Uh_w, 101, Uh_b, Wsh_b, tid);
  // ---- stage xt tile = [x | t | 1 | 0-pad] as f16, row-major ----
  for (int idx = tid; idx < BROWS * HIDN; idx += TPB) {
    int r = idx >> 7, c = idx & 127;
    int grow = rowBase + r;
    float v = (c < XDIM) ? x[grow * XDIM + c]
            : ((c == XDIM) ? t[grow] : ((c == XDIM + 1) ? 1.0f : 0.0f));
    sbuf[idx] = (_Float16)v;
  }
  __syncthreads();

  _Float16* swave  = sbuf  + wave * 16 * HIDN;   // this wave's 16 rows
  const _Float16* srow  = swave  + ln * HIDN;
  _Float16* srwave = srbuf + wave * 16 * HIDN;
  const _Float16* srrow = srwave + ln * HIDN;

  // xt A-fragments (4 k-steps of 32)
  v16h xf[4];
#pragma unroll
  for (int ks = 0; ks < 4; ++ks) xf[ks] = load_afrag(srow, ks, g);

  h8 uzP[8], ugP[8], urP[8], uhP[8], sP[8];

  auto project = [&](const _Float16* W, h8* dst) {
#pragma unroll
    for (int nt = 0; nt < 8; ++nt) {
      v8f c = {};   // biases ride in weight row 101 against xt's 1-column
#pragma unroll
      for (int ks = 0; ks < 4; ++ks) {
        v16h bf = *(const v16h*)(W + (((ks << 3) | nt) * 32 + lane) * 16);
        c = wmma_f16(xf[ks], bf, c);
      }
#pragma unroll
      for (int i = 0; i < 8; ++i) dst[nt][i] = (_Float16)c[i];
    }
  };
  project(wbuf + 1 * 16384, uzP);   // uz = xt@Uz + Uz_b + Wsz_b
  project(wbuf + 2 * 16384, ugP);
  project(wbuf + 3 * 16384, urP);
  project(wbuf + 4 * 16384, uhP);

  // S1 = tanh(xt @ Sw + Sw_b); keep C-layout copy + write row-major to LDS
#pragma unroll
  for (int nt = 0; nt < 8; ++nt) {
    v8f c = {};
#pragma unroll
    for (int ks = 0; ks < 4; ++ks) {
      v16h bf = *(const v16h*)(wbuf + (((ks << 3) | nt) * 32 + lane) * 16);
      c = wmma_f16(xf[ks], bf, c);
    }
#pragma unroll
    for (int i = 0; i < 8; ++i) {
      float sv = fast_tanh(c[i]);
      sP[nt][i] = (_Float16)sv;
      swave[(i + (g << 3)) * HIDN + nt * 16 + ln] = (_Float16)sv;
    }
  }
  __syncthreads();   // all waves done with U weights

  // ---- stage recurrent weights over slots 0..3 ----
  stage_weight(wbuf + 0 * 16384, Wsz_w, 128, nullptr, nullptr, tid);
  stage_weight(wbuf + 1 * 16384, Wsg_w, 128, nullptr, nullptr, tid);
  stage_weight(wbuf + 2 * 16384, Wsr_w, 128, nullptr, nullptr, tid);
  stage_weight(wbuf + 3 * 16384, Wsh_w, 128, nullptr, nullptr, tid);
  __syncthreads();

  const _Float16* Wz2 = wbuf + 0 * 16384;
  const _Float16* Wg2 = wbuf + 1 * 16384;
  const _Float16* Wr2 = wbuf + 2 * 16384;
  const _Float16* Wh2 = wbuf + 3 * 16384;

  const int nl = *nlp;
  for (int layer = 1; layer < nl; ++layer) {
    v16h sF[4];
#pragma unroll
    for (int ks = 0; ks < 4; ++ks) sF[ks] = load_afrag(srow, ks, g);

    h8 zP[8], gP[8];
    // Z = tanh(uz + S@Wsz)
#pragma unroll
    for (int nt = 0; nt < 8; ++nt) {
      v8f c;
#pragma unroll
      for (int i = 0; i < 8; ++i) c[i] = (float)uzP[nt][i];
#pragma unroll
      for (int ks = 0; ks < 4; ++ks)
        c = wmma_f16(sF[ks], *(const v16h*)(Wz2 + (((ks << 3) | nt) * 32 + lane) * 16), c);
#pragma unroll
      for (int i = 0; i < 8; ++i) zP[nt][i] = (_Float16)fast_tanh(c[i]);
    }
    // G = tanh(ug + S@Wsg)
#pragma unroll
    for (int nt = 0; nt < 8; ++nt) {
      v8f c;
#pragma unroll
      for (int i = 0; i < 8; ++i) c[i] = (float)ugP[nt][i];
#pragma unroll
      for (int ks = 0; ks < 4; ++ks)
        c = wmma_f16(sF[ks], *(const v16h*)(Wg2 + (((ks << 3) | nt) * 32 + lane) * 16), c);
#pragma unroll
      for (int i = 0; i < 8; ++i) gP[nt][i] = (_Float16)fast_tanh(c[i]);
    }
    // R = tanh(ur + S@Wsr); SR = S*R -> row-major LDS
#pragma unroll
    for (int nt = 0; nt < 8; ++nt) {
      v8f c;
#pragma unroll
      for (int i = 0; i < 8; ++i) c[i] = (float)urP[nt][i];
#pragma unroll
      for (int ks = 0; ks < 4; ++ks)
        c = wmma_f16(sF[ks], *(const v16h*)(Wr2 + (((ks << 3) | nt) * 32 + lane) * 16), c);
#pragma unroll
      for (int i = 0; i < 8; ++i) {
        float rv  = fast_tanh(c[i]);
        float srv = rv * (float)sP[nt][i];
        srwave[(i + (g << 3)) * HIDN + nt * 16 + ln] = (_Float16)srv;
      }
    }
    v16h srF[4];
#pragma unroll
    for (int ks = 0; ks < 4; ++ks) srF[ks] = load_afrag(srrow, ks, g);

    // H = uh + SR@Wsh ; newS = (1-G)*H + Z*S
#pragma unroll
    for (int nt = 0; nt < 8; ++nt) {
      v8f c;
#pragma unroll
      for (int i = 0; i < 8; ++i) c[i] = (float)uhP[nt][i];
#pragma unroll
      for (int ks = 0; ks < 4; ++ks)
        c = wmma_f16(srF[ks], *(const v16h*)(Wh2 + (((ks << 3) | nt) * 32 + lane) * 16), c);
#pragma unroll
      for (int i = 0; i < 8; ++i) {
        float hv = c[i];
        float zv = (float)zP[nt][i], gv = (float)gP[nt][i], sv = (float)sP[nt][i];
        float ns = (1.0f - gv) * hv + zv * sv;
        sP[nt][i] = (_Float16)ns;
        swave[(i + (g << 3)) * HIDN + nt * 16 + ln] = (_Float16)ns;
      }
    }
  }

  // ---- out = S @ Wf + Wf_b  (h8-vectorized dot + cross-half shuffle) ----
  float wfb = Wf_b[0];
  float acc = 0.0f;
  const _Float16* sr2 = swave + ln * HIDN + g * 64;
  const _Float16* wfp = wfh + g * 64;
#pragma unroll
  for (int c8 = 0; c8 < 8; ++c8) {
    h8 sv = *(const h8*)(sr2 + c8 * 8);
    h8 wv = *(const h8*)(wfp + c8 * 8);
#pragma unroll
    for (int i = 0; i < 8; ++i) acc += (float)sv[i] * (float)wv[i];
  }
  acc += __shfl_xor(acc, 16, 32);
  if (g == 0) out[rowBase + wave * 16 + ln] = acc + wfb;
}

extern "C" void kernel_launch(void* const* d_in, const int* in_sizes, int n_in,
                              void* d_out, int out_size, void* d_ws, size_t ws_size,
                              hipStream_t stream) {
  (void)n_in; (void)out_size; (void)d_ws; (void)ws_size;
  const float* x     = (const float*)d_in[0];
  const float* t     = (const float*)d_in[1];
  const float* Sw_w  = (const float*)d_in[2];
  const float* Sw_b  = (const float*)d_in[3];
  const float* Uz_w  = (const float*)d_in[4];
  const float* Uz_b  = (const float*)d_in[5];
  const float* Wsz_w = (const float*)d_in[6];
  const float* Wsz_b = (const float*)d_in[7];
  const float* Ug_w  = (const float*)d_in[8];
  const float* Ug_b  = (const float*)d_in[9];
  const float* Wsg_w = (const float*)d_in[10];
  const float* Wsg_b = (const float*)d_in[11];
  const float* Ur_w  = (const float*)d_in[12];
  const float* Ur_b  = (const float*)d_in[13];
  const float* Wsr_w = (const float*)d_in[14];
  const float* Wsr_b = (const float*)d_in[15];
  const float* Uh_w  = (const float*)d_in[16];
  const float* Uh_b  = (const float*)d_in[17];
  const float* Wsh_w = (const float*)d_in[18];
  const float* Wsh_b = (const float*)d_in[19];
  const float* Wf_w  = (const float*)d_in[20];
  const float* Wf_b  = (const float*)d_in[21];
  const int*   nlp   = (const int*)d_in[22];
  float* out = (float*)d_out;

  int Bn = in_sizes[1];                 // t has B elements
  int blocks = Bn / BROWS;
  size_t shmem = 196608 + 128 * sizeof(_Float16);   // 196864 B dynamic LDS

  hipFuncSetAttribute((const void*)dgm_fused_kernel,
                      hipFuncAttributeMaxDynamicSharedMemorySize, (int)shmem);
  dgm_fused_kernel<<<blocks, TPB, shmem, stream>>>(
      x, t, Sw_w, Sw_b, Uz_w, Uz_b, Wsz_w, Wsz_b, Ug_w, Ug_b, Wsg_w, Wsg_b,
      Ur_w, Ur_b, Wsr_w, Wsr_b, Uh_w, Uh_b, Wsh_w, Wsh_b, Wf_w, Wf_b, nlp, out);
}
